// HierAttNet_901943132712
// MI455X (gfx1250) — compile-verified, matrix operand order checked by reading, not run on hardware
//
#include <hip/hip_runtime.h>
#include <hip/hip_bf16.h>
#include <math.h>

typedef __attribute__((ext_vector_type(16))) _Float16 v16h;
typedef __attribute__((ext_vector_type(8)))  _Float16 v8h;
typedef __attribute__((ext_vector_type(8)))  float    v8f;
typedef __attribute__((ext_vector_type(4)))  float    v4f;

#define NB    8
#define ND    1024
#define DREAL 100
#define DPAD  128
#define NV    30000
#define NC    512
#define NTILE (NV / 16)   // 1875 v-tiles of 16

// ---------------------------------------------------------------------------
// Prep 1: repack Vv_T [D, Nv] (f32, row = dim k) -> Bh [Nv][DPAD] f16, zero pad
// ---------------------------------------------------------------------------
__global__ void prep_B_kernel(const float* __restrict__ VvT, _Float16* __restrict__ Bh) {
    int v = blockIdx.x * blockDim.x + threadIdx.x;   // coalesced read in v
    int k = blockIdx.y;                              // 0..127
    if (v >= NV) return;
    float x = (k < DREAL) ? VvT[(size_t)k * NV + v] : 0.0f;
    Bh[(size_t)v * DPAD + k] = (_Float16)x;
}

// ---------------------------------------------------------------------------
// Prep 2: gather Ah [b][d][DPAD] f16 = emb[doc_index[b,d]][k], zero pad
// ---------------------------------------------------------------------------
__global__ void prep_A_kernel(const int* __restrict__ doc_index,
                              const float* __restrict__ emb,
                              _Float16* __restrict__ Ah) {
    int idx = blockIdx.x * blockDim.x + threadIdx.x;   // over 8*1024*128
    if (idx >= NB * ND * DPAD) return;
    int k  = idx & (DPAD - 1);
    int bd = idx >> 7;                                 // b*1024 + d
    int row = doc_index[bd];
    float x = (k < DREAL) ? emb[(size_t)row * DREAL + k] : 0.0f;
    Ah[idx] = (_Float16)x;
}

// ---------------------------------------------------------------------------
// Main fused kernel: sim = Ah x Bh^T via v_wmma_f32_16x16x32_f16, then
// uniform-bin digitize -> bin-weight LDS lookup -> attn-weighted d-reduction.
// Block = 128 threads (4 waves); wave w handles v-tile (blockIdx.x*4 + w).
// ---------------------------------------------------------------------------
__global__ void __launch_bounds__(128)
wmma_bin_reduce_kernel(const _Float16* __restrict__ Ah,
                       const _Float16* __restrict__ Bh,
                       const float* __restrict__ attn,
                       const float* __restrict__ bin_diff,
                       const float* __restrict__ bin_start,
                       float* __restrict__ Wt) {
    __shared__ float binw[16];
    if (threadIdx.x == 0) {
        float acc = bin_start[0];
        #pragma unroll
        for (int k = 0; k < 16; ++k) {
            float dk = bin_diff[k];
            acc += (dk > 0.0f) ? dk : 0.0f;   // cumsum(relu(diff))
            binw[k] = acc;
        }
    }
    __syncthreads();

    const int b    = blockIdx.y;
    const int wave = threadIdx.x >> 5;
    const int lane = threadIdx.x & 31;
    const int tile = blockIdx.x * 4 + wave;
    if (tile >= NTILE) return;                 // wave-uniform: EXEC stays all-1s

    const int v0   = tile * 16;
    const int n    = lane & 15;                // row (A) / col (B,C) owned by lane
    const int koff = (lane >> 4) << 3;         // 0 or 8 (K-half select, ISA 7.12.2)

    const _Float16* arowbase = Ah + (size_t)b * ND * DPAD;
    const _Float16* brow     = Bh + (size_t)(v0 + n) * DPAD;
    const float*    attb     = attn + b * ND;

    const float invh = 14.0f / 1.49f;          // uniform bin_mid spacing^-1
    float wacc = 0.0f;

    for (int d0 = 0; d0 < ND; d0 += 16) {
        const _Float16* arow = arowbase + (size_t)(d0 + n) * DPAD;
        __builtin_prefetch(arow + 16 * DPAD, 0, 1);   // next d-tile row

        v8f c = {};
        #pragma unroll
        for (int kk = 0; kk < DPAD; kk += 32) {
            v8h alo = *(const v8h*)(arow + kk + koff);
            v8h ahi = *(const v8h*)(arow + kk + koff + 16);
            v8h blo = *(const v8h*)(brow + kk + koff);
            v8h bhi = *(const v8h*)(brow + kk + koff + 16);
            v16h a  = __builtin_shufflevector(alo, ahi, 0,1,2,3,4,5,6,7,8,9,10,11,12,13,14,15);
            v16h bm = __builtin_shufflevector(blo, bhi, 0,1,2,3,4,5,6,7,8,9,10,11,12,13,14,15);
            c = __builtin_amdgcn_wmma_f32_16x16x32_f16(
                    false, a, false, bm, (short)0, c, false, false);
        }

        // C layout: VGPR j holds M = j (lanes 0-15) or j+8 (lanes 16-31); N = lane&15
        const float* ap = attb + d0 + koff;    // koff == C-row offset for this half
        v4f a0 = *(const v4f*)(ap);
        v4f a1 = *(const v4f*)(ap + 4);
        float av[8] = {a0.x, a0.y, a0.z, a0.w, a1.x, a1.y, a1.z, a1.w};

        #pragma unroll
        for (int j = 0; j < 8; ++j) {
            float sim = c[j];
            // digit = #(bin_mid <= sim), bins uniform on [-0.5, 0.99]
            int di = (int)floorf((sim + 0.5f) * invh) + 1;
            di = di < 0 ? 0 : (di > 15 ? 15 : di);
            wacc = fmaf(binw[di], av[j], wacc);
        }
    }

    // fold the two lane halves (same v column, rows 0-7 vs 8-15)
    wacc += __shfl_xor(wacc, 16, 32);
    if (lane < 16) Wt[(size_t)b * NV + v0 + n] = wacc;
}

// ---------------------------------------------------------------------------
// Final: out[b,n] = dot(Wt[b,:], phi[n,:]) / max(L1(phi[n,:]), 1e-12)
// One block per n; single pass over phi row (memory floor).
// ---------------------------------------------------------------------------
__global__ void __launch_bounds__(256)
final_kernel(const float* __restrict__ phi,
             const float* __restrict__ Wt,
             float* __restrict__ out) {
    const int n = blockIdx.x;
    const int t = threadIdx.x;
    const float* pr = phi + (size_t)n * NV;

    float asum = 0.0f;
    float dot[NB];
    #pragma unroll
    for (int b = 0; b < NB; ++b) dot[b] = 0.0f;

    for (int v = t; v < NV; v += 256) {
        float p = pr[v];
        asum += fabsf(p);
        #pragma unroll
        for (int b = 0; b < NB; ++b)
            dot[b] = fmaf(Wt[(size_t)b * NV + v], p, dot[b]);
    }

    __shared__ float red[256];
    __shared__ float tot[NB + 1];
    #pragma unroll
    for (int q = 0; q < NB + 1; ++q) {
        red[t] = (q == 0) ? asum : dot[q - 1];
        __syncthreads();
        for (int s = 128; s > 0; s >>= 1) {
            if (t < s) red[t] += red[t + s];
            __syncthreads();
        }
        if (t == 0) tot[q] = red[0];
        __syncthreads();
    }

    if (t == 0) {
        float inv = 1.0f / fmaxf(tot[0], 1e-12f);
        #pragma unroll
        for (int b = 0; b < NB; ++b)
            out[(size_t)b * NC + n] = tot[b + 1] * inv;
    }
}

// ---------------------------------------------------------------------------
extern "C" void kernel_launch(void* const* d_in, const int* in_sizes, int n_in,
                              void* d_out, int out_size, void* d_ws, size_t ws_size,
                              hipStream_t stream) {
    (void)in_sizes; (void)n_in; (void)out_size; (void)ws_size;

    const int*   doc_index = (const int*)  d_in[0];   // [8,1024]
    const float* attn      = (const float*)d_in[1];   // [8,1024]
    const float* emb       = (const float*)d_in[2];   // [30001,100]
    const float* VvT       = (const float*)d_in[3];   // [100,30000]
    const float* phi       = (const float*)d_in[4];   // [512,30000]
    const float* bin_diff  = (const float*)d_in[5];   // [16]
    const float* bin_start = (const float*)d_in[6];   // [1]
    float*       out       = (float*)d_out;           // [8,512]

    // workspace layout (all 16B aligned)
    _Float16* Bh = (_Float16*)d_ws;                         // 30000*128 f16 = 7.68 MB
    _Float16* Ah = Bh + (size_t)NV * DPAD;                  // 8*1024*128 f16 = 2.10 MB
    float*    Wt = (float*)(Ah + (size_t)NB * ND * DPAD);   // 8*30000 f32 = 0.96 MB

    // prep: repack B operand
    {
        dim3 grid((NV + 255) / 256, DPAD);
        prep_B_kernel<<<grid, 256, 0, stream>>>(VvT, Bh);
    }
    // prep: gather + convert A operand
    {
        int total = NB * ND * DPAD;
        prep_A_kernel<<<(total + 255) / 256, 256, 0, stream>>>(doc_index, emb, Ah);
    }
    // main fused WMMA + digitize + attn reduction
    {
        dim3 grid((NTILE + 3) / 4, NB);   // 469 x 8 blocks, 4 waves each
        wmma_bin_reduce_kernel<<<grid, 128, 0, stream>>>(
            Ah, Bh, attn, bin_diff, bin_start, Wt);
    }
    // final GEMM vs L1-normalized phi
    final_kernel<<<NC, 256, 0, stream>>>(phi, Wt, out);
}
